// ImageCaptioner_67963562492488
// MI455X (gfx1250) — matrix operand order repaired and checked by reading, not compiled, not run on hardware
//
#include <hip/hip_runtime.h>

#define BN 32
#define FIMG 2048
#define VV 12000
#define DWE 300
#define HH 512
#define KB 3
#define TT 32
#define NR 96            // BN*KB
#define KTOT 2880        // (300 + 2048 + 512) padded to multiple of 32
#define KT_Z 90          // KTOT/32
#define KT_H 16          // HH/32
#define ZN 2048          // 4*HH
#define NEGINF (-1e9f)
#define START_ID 1
#define END_ID 2

typedef __attribute__((ext_vector_type(16))) __bf16 v16bf;
typedef __attribute__((ext_vector_type(8)))  float  v8f;

__device__ inline __bf16 f2bf(float f) {
  union { float f; unsigned u; } v; v.f = f;
  unsigned r = (v.u + 0x7fffu + ((v.u >> 16) & 1u)) >> 16;
  unsigned short s = (unsigned short)r;
  __bf16 o; __builtin_memcpy(&o, &s, 2);
  return o;
}

__device__ inline float sigm(float x) { return 1.0f / (1.0f + __expf(-x)); }

// =====================================================================
// Fragment packing.  Fragment layout (per 16x16x32 WMMA K-tile):
//   A frag: [mt][kt][lane][16 bf16]  (lane: M=lane&15, K pairs per VGPR, +8K for upper half)
//   B frag: [nt][kt][lane][16 bf16]  (lane: N=lane&15, K pairs per VGPR, +8K for upper half)
// Each lane's 16 bf16 = 32 contiguous bytes -> two global_load_b128.
// =====================================================================

// pack fused [Wx ; Wh ; 0] (K x 2048) into B fragments, ktiles = KT_Z
__global__ void k_pack_wall(const float* __restrict__ Wx, const float* __restrict__ Wh,
                            __bf16* __restrict__ out) {
  int lane = threadIdx.x & 31;
  int kt = blockIdx.x;                 // 0..KT_Z-1
  int nt = blockIdx.y;                 // 0..127
  int half = lane >> 4, l16 = lane & 15;
  int n = nt * 16 + l16;
  __bf16* dst = out + ((size_t)(nt * KT_Z + kt) * 32 + lane) * 16;
#pragma unroll
  for (int i = 0; i < 8; ++i) {
#pragma unroll
    for (int d = 0; d < 2; ++d) {
      int k = kt * 32 + (i < 4 ? 8 * half + 2 * i : 16 + 8 * half + 2 * (i - 4)) + d;
      float v = 0.0f;
      if (k < DWE + FIMG)           v = Wx[(size_t)k * ZN + n];
      else if (k < DWE + FIMG + HH) v = Wh[(size_t)(k - DWE - FIMG) * ZN + n];
      dst[2 * i + d] = f2bf(v);
    }
  }
}

// pack generic fp32 B (Kreal x N row-major) into fragments
__global__ void k_pack_b(const float* __restrict__ W, int N, int Kreal, int ktiles,
                         __bf16* __restrict__ out) {
  int lane = threadIdx.x & 31;
  int kt = blockIdx.x;
  int nt = blockIdx.y;
  int half = lane >> 4, l16 = lane & 15;
  int n = nt * 16 + l16;
  __bf16* dst = out + ((size_t)(nt * ktiles + kt) * 32 + lane) * 16;
#pragma unroll
  for (int i = 0; i < 8; ++i) {
#pragma unroll
    for (int d = 0; d < 2; ++d) {
      int k = kt * 32 + (i < 4 ? 8 * half + 2 * i : 16 + 8 * half + 2 * (i - 4)) + d;
      float v = (k < Kreal) ? W[(size_t)k * N + n] : 0.0f;
      dst[2 * i + d] = f2bf(v);
    }
  }
}

// pack generic fp32 A (M x lda row-major) into fragments
__global__ void k_pack_a(const float* __restrict__ A, int lda, int ktiles,
                         __bf16* __restrict__ out) {
  int lane = threadIdx.x & 31;
  int kt = blockIdx.x;
  int mt = blockIdx.y;
  int half = lane >> 4, l16 = lane & 15;
  const float* Ar = A + (size_t)(mt * 16 + l16) * lda + kt * 32;
  __bf16* dst = out + ((size_t)(mt * ktiles + kt) * 32 + lane) * 16;
#pragma unroll
  for (int i = 0; i < 4; ++i) {
    int k0 = 8 * half + 2 * i;
    dst[2 * i]           = f2bf(Ar[k0]);
    dst[2 * i + 1]       = f2bf(Ar[k0 + 1]);
    int k1 = 16 + k0;
    dst[2 * (i + 4)]     = f2bf(Ar[k1]);
    dst[2 * (i + 4) + 1] = f2bf(Ar[k1 + 1]);
  }
}

// build + pack concat [emb | img | h | 0] A-fragments directly (ktiles = KT_Z)
__global__ void k_pack_xh(const float* __restrict__ E, const float* __restrict__ img,
                          const float* __restrict__ h, const int* __restrict__ tok,
                          int tok_stride, __bf16* __restrict__ out) {
  int lane = threadIdx.x & 31;
  int kt = blockIdx.x;                 // 0..KT_Z-1
  int mt = blockIdx.y;                 // 0..5
  int half = lane >> 4, l16 = lane & 15;
  int r = mt * 16 + l16;
  int e = tok[r * tok_stride];
  const float* erow = E + (size_t)e * DWE;
  const float* irow = img + (size_t)(r / KB) * FIMG;
  const float* hrow = h + (size_t)r * HH;
  __bf16* dst = out + ((size_t)(mt * KT_Z + kt) * 32 + lane) * 16;
  auto fetch = [&](int k) -> float {
    if (k < DWE)                  return erow[k];
    if (k < DWE + FIMG)           return irow[k - DWE];
    if (k < DWE + FIMG + HH)      return hrow[k - DWE - FIMG];
    return 0.0f;
  };
#pragma unroll
  for (int i = 0; i < 4; ++i) {
    int k0 = kt * 32 + 8 * half + 2 * i;
    dst[2 * i]           = f2bf(fetch(k0));
    dst[2 * i + 1]       = f2bf(fetch(k0 + 1));
    int k1 = k0 + 16;
    dst[2 * (i + 4)]     = f2bf(fetch(k1));
    dst[2 * (i + 4) + 1] = f2bf(fetch(k1 + 1));
  }
}

// =====================================================================
// WMMA GEMM on packed fragments.  One wave handles a 16 x (NT*16) output
// strip; A fragment reused across NT back-to-back v_wmma ops per K-tile.
// grid.x = Ntiles/NT, grid.y = Mtiles, block = 32.
// =====================================================================

template <int NT>
__global__ void gemm_pk(const v16bf* __restrict__ Apk, const v16bf* __restrict__ Bpk,
                        float* __restrict__ C, int ldc, const float* __restrict__ bias,
                        int ktiles) {
  int lane = threadIdx.x & 31;
  int half = lane >> 4, l16 = lane & 15;
  int mt = blockIdx.y;
  int nt0 = blockIdx.x * NT;
  v8f acc[NT];
#pragma unroll
  for (int j = 0; j < NT; ++j) acc[j] = (v8f){0.f, 0.f, 0.f, 0.f, 0.f, 0.f, 0.f, 0.f};
  const v16bf* Ait = Apk + (size_t)mt * ktiles * 32 + lane;
  for (int kt = 0; kt < ktiles; ++kt) {
    v16bf a = Ait[(size_t)kt * 32];
#pragma unroll
    for (int j = 0; j < NT; ++j) {
      v16bf b = Bpk[((size_t)(nt0 + j) * ktiles + kt) * 32 + lane];
      acc[j] = __builtin_amdgcn_wmma_f32_16x16x32_bf16(false, a, false, b, (short)0,
                                                       acc[j], false, false);
    }
  }
#pragma unroll
  for (int j = 0; j < NT; ++j) {
    int n = (nt0 + j) * 16 + l16;
    float bb = bias ? bias[n] : 0.0f;
#pragma unroll
    for (int r = 0; r < 8; ++r) {
      int mm = mt * 16 + r + 8 * half;     // C/D layout: VGPR r -> M = r + 8*half
      C[(size_t)mm * ldc + n] = acc[j][r] + bb;
    }
  }
}

// ---------------- LSTM cells ----------------

__global__ void k_lstm(const float* __restrict__ z, const float* __restrict__ cin,
                       float* __restrict__ h2, float* __restrict__ c2) {
  int idx = blockIdx.x * blockDim.x + threadIdx.x;
  if (idx >= NR * HH) return;
  int r = idx / HH, j = idx % HH;
  const float* zr = z + (size_t)r * ZN;
  float i = zr[j], f = zr[HH + j], g = zr[2 * HH + j], o = zr[3 * HH + j];
  float cn = sigm(f + 1.0f) * cin[idx] + sigm(i) * tanhf(g);
  float hn = sigm(o) * tanhf(cn);
  c2[idx] = cn;
  h2[idx] = hn;
}

__global__ void k_lstm_teacher(const float* __restrict__ z, float* __restrict__ h,
                               float* __restrict__ c, const int* __restrict__ len,
                               int t, float* __restrict__ acts) {
  int idx = blockIdx.x * blockDim.x + threadIdx.x;
  if (idx >= NR * HH) return;
  int r = idx / HH, j = idx % HH;
  const float* zr = z + (size_t)r * ZN;
  float i = zr[j], f = zr[HH + j], g = zr[2 * HH + j], o = zr[3 * HH + j];
  float cn = sigm(f + 1.0f) * c[idx] + sigm(i) * tanhf(g);
  float hn = sigm(o) * tanhf(cn);
  float* arow = acts + ((size_t)r * (TT + 1) + t) * HH;
  if (t < len[r]) {
    c[idx] = cn;
    h[idx] = hn;
    arow[j] = hn;
  } else {
    arow[j] = 0.0f;
  }
}

// ---------------- log-softmax (per row of lp[NR,VV]) ----------------

__global__ void k_logsoftmax(float* __restrict__ lp) {
  int r = blockIdx.x;
  float* row = lp + (size_t)r * VV;
  __shared__ float red[256];
  int tid = threadIdx.x;
  float m = -3.4e38f;
  for (int i = tid; i < VV; i += 256) m = fmaxf(m, row[i]);
  red[tid] = m; __syncthreads();
  for (int s = 128; s > 0; s >>= 1) {
    if (tid < s) red[tid] = fmaxf(red[tid], red[tid + s]);
    __syncthreads();
  }
  m = red[0]; __syncthreads();
  float sum = 0.0f;
  for (int i = tid; i < VV; i += 256) sum += __expf(row[i] - m);
  red[tid] = sum; __syncthreads();
  for (int s = 128; s > 0; s >>= 1) {
    if (tid < s) red[tid] += red[tid + s];
    __syncthreads();
  }
  float lse = m + __logf(red[0]);
  __syncthreads();
  for (int i = tid; i < VV; i += 256) row[i] -= lse;
}

// ---------------- beam top-K + state update (one block per batch) ----------------

__global__ void k_topk(const float* __restrict__ lp, float* __restrict__ beam,
                       int* __restrict__ prev, int* __restrict__ fin,
                       int* __restrict__ len, const int* __restrict__ ids_in,
                       int* __restrict__ ids_out, int* __restrict__ parent, int t) {
  int b = blockIdx.x;
  int tid = threadIdx.x; // 128 threads
  __shared__ float sv[128 * 3];
  __shared__ int   si[128 * 3];
  float lv0 = -3.4e38f, lv1 = -3.4e38f, lv2 = -3.4e38f;
  int   li0 = 0, li1 = 0, li2 = 0;
  for (int i = tid; i < KB * VV; i += 128) {
    int kk = i / VV, w = i % VV;
    int r = b * KB + kk;
    float s = fin[r] ? (beam[r] + ((w == END_ID) ? 0.0f : NEGINF))
                     : (beam[r] + lp[(size_t)r * VV + w]);
    if (s > lv2) {
      if (s > lv0)      { lv2 = lv1; li2 = li1; lv1 = lv0; li1 = li0; lv0 = s; li0 = i; }
      else if (s > lv1) { lv2 = lv1; li2 = li1; lv1 = s; li1 = i; }
      else              { lv2 = s; li2 = i; }
    }
  }
  sv[tid * 3 + 0] = lv0; si[tid * 3 + 0] = li0;
  sv[tid * 3 + 1] = lv1; si[tid * 3 + 1] = li1;
  sv[tid * 3 + 2] = lv2; si[tid * 3 + 2] = li2;
  __syncthreads();
  if (tid == 0) {
    float bv0 = -3.4e38f, bv1 = -3.4e38f, bv2 = -3.4e38f;
    int   bi0 = 0, bi1 = 0, bi2 = 0;
    for (int i = 0; i < 128 * 3; ++i) {
      float s = sv[i]; int ix = si[i];
      if (s > bv2 || (s == bv2 && ix < bi2)) {
        if (s > bv0 || (s == bv0 && ix < bi0))      { bv2 = bv1; bi2 = bi1; bv1 = bv0; bi1 = bi0; bv0 = s; bi0 = ix; }
        else if (s > bv1 || (s == bv1 && ix < bi1)) { bv2 = bv1; bi2 = bi1; bv1 = s; bi1 = ix; }
        else                                        { bv2 = s; bi2 = ix; }
      }
    }
    float sc[3] = {bv0, bv1, bv2};
    int   ix[3] = {bi0, bi1, bi2};
    int par[3], wrd[3], finp[3], lenp[3];
    for (int k = 0; k < KB; ++k) {
      par[k] = ix[k] / VV;
      wrd[k] = ix[k] % VV;
      finp[k] = fin[b * KB + par[k]];
      lenp[k] = len[b * KB + par[k]];
    }
    for (int k = 0; k < KB; ++k) {
      int r = b * KB + k;
      beam[r]   = sc[k];
      prev[r]   = wrd[k];
      parent[r] = par[k];
      for (int tt = 0; tt < TT; ++tt)
        ids_out[r * TT + tt] = ids_in[(b * KB + par[k]) * TT + tt];
      ids_out[r * TT + t] = wrd[k];
      len[r] = finp[k] ? lenp[k] : lenp[k] + 1;
      fin[r] = finp[k] | (wrd[k] == END_ID);
    }
  }
}

__global__ void k_gather(const float* __restrict__ h2, const float* __restrict__ c2,
                         const int* __restrict__ parent, float* __restrict__ h,
                         float* __restrict__ c) {
  int idx = blockIdx.x * blockDim.x + threadIdx.x;
  if (idx >= NR * HH) return;
  int r = idx / HH, j = idx % HH;
  int src = (r / KB) * KB + parent[r];
  h[idx] = h2[(size_t)src * HH + j];
  c[idx] = c2[(size_t)src * HH + j];
}

// ---------------- misc ----------------

__global__ void k_init_beam(float* beam, int* prev, int* fin, int* len, int* ids) {
  int tid = threadIdx.x;
  for (int i = tid; i < NR; i += blockDim.x) {
    beam[i] = (i % KB == 0) ? 0.0f : NEGINF;
    prev[i] = START_ID;
    fin[i] = 0;
    len[i] = 0;
  }
  for (int i = tid; i < NR * TT; i += blockDim.x) ids[i] = 0;
}

__global__ void k_zero(float* a, int n) {
  int idx = blockIdx.x * blockDim.x + threadIdx.x;
  if (idx < n) a[idx] = 0.0f;
}

__global__ void k_build_seq(const int* __restrict__ ids, int* __restrict__ seq) {
  int idx = blockIdx.x * blockDim.x + threadIdx.x;
  if (idx >= NR * (TT + 1)) return;
  int r = idx / (TT + 1), t = idx % (TT + 1);
  seq[idx] = (t == 0) ? START_ID : ids[r * TT + (t - 1)];
}

__global__ void k_argmax(const float* __restrict__ logits, int* __restrict__ out) {
  int r = blockIdx.x;
  const float* row = logits + (size_t)r * VV;
  __shared__ float sv[256];
  __shared__ int   si[256];
  int tid = threadIdx.x;
  float bv = -3.4e38f; int bi = 0;
  for (int i = tid; i < VV; i += 256) {
    float v = row[i];
    if (v > bv) { bv = v; bi = i; }
  }
  sv[tid] = bv; si[tid] = bi; __syncthreads();
  for (int s = 128; s > 0; s >>= 1) {
    if (tid < s) {
      if (sv[tid + s] > sv[tid] || (sv[tid + s] == sv[tid] && si[tid + s] < si[tid])) {
        sv[tid] = sv[tid + s]; si[tid] = si[tid + s];
      }
    }
    __syncthreads();
  }
  if (tid == 0) out[r] = si[0];
}

// ---------------- host orchestration ----------------

extern "C" void kernel_launch(void* const* d_in, const int* in_sizes, int n_in,
                              void* d_out, int out_size, void* d_ws, size_t ws_size,
                              hipStream_t stream) {
  (void)in_sizes; (void)n_in; (void)out_size; (void)ws_size;
  const float* img = (const float*)d_in[0];   // [32,2048]
  const float* E   = (const float*)d_in[1];   // [12000,300]
  const float* Wx  = (const float*)d_in[2];   // [2348,2048]
  const float* Wh  = (const float*)d_in[3];   // [512,2048]
  const float* bl  = (const float*)d_in[4];   // [2048]
  const float* Wl  = (const float*)d_in[5];   // [512,12000]
  const float* bo  = (const float*)d_in[6];   // [12000]

  char* p = (char*)d_ws;
  auto carve = [&](size_t bytes) {
    char* r = p;
    p += (bytes + 255) & ~(size_t)255;
    return (void*)r;
  };
  // packed weight fragments (built once per launch; L2-resident afterwards)
  v16bf* Wall_pk = (v16bf*)carve((size_t)128 * KT_Z * 32 * 32);          // [nt][kt][lane][16]bf16
  v16bf* Wl_pk   = (v16bf*)carve((size_t)(VV / 16) * KT_H * 32 * 32);
  // packed activation fragments
  v16bf* xh_pk   = (v16bf*)carve((size_t)6 * KT_Z * 32 * 32);
  v16bf* h2_pk   = (v16bf*)carve((size_t)6 * KT_H * 32 * 32);
  v16bf* acts_pk = (v16bf*)carve((size_t)198 * KT_H * 32 * 32);
  // fp32 state
  float* z    = (float*)carve((size_t)NR * ZN * 4);
  float* h    = (float*)carve((size_t)NR * HH * 4);
  float* c    = (float*)carve((size_t)NR * HH * 4);
  float* h2   = (float*)carve((size_t)NR * HH * 4);
  float* c2   = (float*)carve((size_t)NR * HH * 4);
  float* lp   = (float*)carve((size_t)NR * VV * 4);
  float* beam = (float*)carve(NR * 4);
  int* prev   = (int*)carve(NR * 4);
  int* fin    = (int*)carve(NR * 4);
  int* len    = (int*)carve(NR * 4);
  int* parent = (int*)carve(NR * 4);
  int* ids_a  = (int*)carve((size_t)NR * TT * 4);
  int* ids_b  = (int*)carve((size_t)NR * TT * 4);
  int* seq    = (int*)carve((size_t)NR * (TT + 1) * 4);
  float* acts = (float*)carve((size_t)NR * (TT + 1) * HH * 4);

  // one-time weight conversion + fragment packing
  k_pack_wall<<<dim3(KT_Z, ZN / 16), 32, 0, stream>>>(Wx, Wh, (__bf16*)Wall_pk);
  k_pack_b<<<dim3(KT_H, VV / 16), 32, 0, stream>>>(Wl, VV, HH, KT_H, (__bf16*)Wl_pk);
  k_init_beam<<<1, 256, 0, stream>>>(beam, prev, fin, len, ids_a);
  k_zero<<<(NR * HH + 255) / 256, 256, 0, stream>>>(h, NR * HH);
  k_zero<<<(NR * HH + 255) / 256, 256, 0, stream>>>(c, NR * HH);

  // ---- beam decode: 32 sequential steps ----
  int* idc = ids_a;
  int* ido = ids_b;
  for (int t = 0; t < TT; ++t) {
    k_pack_xh<<<dim3(KT_Z, NR / 16), 32, 0, stream>>>(E, img, h, prev, 1,
                                                      (__bf16*)xh_pk);
    gemm_pk<4><<<dim3((ZN / 16) / 4, NR / 16), 32, 0, stream>>>(xh_pk, Wall_pk, z, ZN,
                                                                bl, KT_Z);
    k_lstm<<<(NR * HH + 255) / 256, 256, 0, stream>>>(z, c, h2, c2);
    k_pack_a<<<dim3(KT_H, NR / 16), 32, 0, stream>>>(h2, HH, KT_H, (__bf16*)h2_pk);
    gemm_pk<5><<<dim3((VV / 16) / 5, NR / 16), 32, 0, stream>>>(h2_pk, Wl_pk, lp, VV,
                                                                bo, KT_H);
    k_logsoftmax<<<NR, 256, 0, stream>>>(lp);
    k_topk<<<BN, 128, 0, stream>>>(lp, beam, prev, fin, len, idc, ido, parent, t);
    { int* tmp = idc; idc = ido; ido = tmp; }
    k_gather<<<(NR * HH + 255) / 256, 256, 0, stream>>>(h2, c2, parent, h, c);
  }

  // ---- teacher-forced RNN: 33 sequential steps ----
  k_build_seq<<<(NR * (TT + 1) + 255) / 256, 256, 0, stream>>>(idc, seq);
  k_zero<<<(NR * HH + 255) / 256, 256, 0, stream>>>(h, NR * HH);
  k_zero<<<(NR * HH + 255) / 256, 256, 0, stream>>>(c, NR * HH);
  for (int t = 0; t <= TT; ++t) {
    k_pack_xh<<<dim3(KT_Z, NR / 16), 32, 0, stream>>>(E, img, h, seq + t, TT + 1,
                                                      (__bf16*)xh_pk);
    gemm_pk<4><<<dim3((ZN / 16) / 4, NR / 16), 32, 0, stream>>>(xh_pk, Wall_pk, z, ZN,
                                                                bl, KT_Z);
    k_lstm_teacher<<<(NR * HH + 255) / 256, 256, 0, stream>>>(z, h, c, len, t, acts);
  }

  // ---- final logits GEMM [3168,512]x[512,12000] + argmax ----
  float* outf = (float*)d_out;
  k_pack_a<<<dim3(KT_H, (NR * (TT + 1)) / 16), 32, 0, stream>>>(acts, HH, KT_H,
                                                                (__bf16*)acts_pk);
  gemm_pk<5><<<dim3((VV / 16) / 5, (NR * (TT + 1)) / 16), 32, 0, stream>>>(
      acts_pk, Wl_pk, outf, VV, bo, KT_H);
  k_argmax<<<NR * (TT + 1), 256, 0, stream>>>(outf,
                                              (int*)(outf + (size_t)NR * (TT + 1) * VV));
}
